// SimpleModel3_44753559224989
// MI455X (gfx1250) — matrix operand order repaired and checked by reading, not compiled, not run on hardware
//
#include <hip/hip_runtime.h>

typedef __bf16 v8bf  __attribute__((ext_vector_type(8)));
typedef __bf16 v16bf __attribute__((ext_vector_type(16)));
typedef float  v8f   __attribute__((ext_vector_type(8)));

#define B_    512
#define NSEQ_ 128
#define SEQ_  256
#define H_    512
#define H2_   256

// ---------------- helpers ----------------

__device__ __forceinline__ __bf16 to_bf16(float f) {
    unsigned u = __builtin_bit_cast(unsigned, f);
    u += 0x7FFFu + ((u >> 16) & 1u);          // round-to-nearest-even
    unsigned short h = (unsigned short)(u >> 16);
    return __builtin_bit_cast(__bf16, h);
}

__device__ __forceinline__ float fast_sigmoid(float x) {
    return 1.0f / (1.0f + __expf(-x));
}
__device__ __forceinline__ float fast_tanh(float x) {
    float e = __expf(-2.0f * x);
    return (1.0f - e) / (1.0f + e);
}

// A-fragment (16x32 bf16): lane<16 holds K{0..7,16..23}, lane>=16 holds K{8..15,24..31}
__device__ __forceinline__ v16bf load_a_frag(const __bf16* base, int half) {
    v8bf lo = *(const v8bf*)(base + 8 * half);
    v8bf hi = *(const v8bf*)(base + 16 + 8 * half);
    return __builtin_shufflevector(lo, hi, 0,1,2,3,4,5,6,7,8,9,10,11,12,13,14,15);
}
// B-fragment (32x16 bf16): lane<16 holds K{0..7,16..23}, lane>=16 holds K{8..15,24..31}
__device__ __forceinline__ v16bf load_b_frag(const __bf16* base, int half) {
    v8bf lo = *(const v8bf*)(base + 8 * half);
    v8bf hi = *(const v8bf*)(base + 16 + 8 * half);
    return __builtin_shufflevector(lo, hi, 0,1,2,3,4,5,6,7,8,9,10,11,12,13,14,15);
}

__device__ __forceinline__ v8f wmma_bf16(v16bf a, v16bf b, v8f c) {
    return __builtin_amdgcn_wmma_f32_16x16x32_bf16(
        false, a, false, b, (short)0, c, false, false);
}

// ---------------- conversion / staging kernels ----------------

// x (B, T, SEQ) f32  ->  xT (T, B, SEQ) bf16
__global__ __launch_bounds__(256) void convert_x_kernel(const float* __restrict__ x,
                                                        __bf16* __restrict__ xT) {
    size_t i = (size_t)blockIdx.x * 256 + threadIdx.x;
    int    k  = (int)(i & (SEQ_ - 1));
    size_t bt = i >> 8;                 // SEQ_=256
    int    t  = (int)(bt & (NSEQ_ - 1));
    size_t b  = bt >> 7;                // NSEQ_=128
    xT[(((size_t)t * B_) + b) * SEQ_ + k] = to_bf16(x[i]);
}

// [Wx (K0,4H); Wh (K1,4H)] f32  ->  WT (4H, K0+K1) bf16 row-major (N-major)
__global__ __launch_bounds__(256) void convert_wpair_kernel(const float* __restrict__ Wx,
                                                            const float* __restrict__ Wh,
                                                            __bf16* __restrict__ WT,
                                                            int K0, int Ktot) {
    size_t i = (size_t)blockIdx.x * 256 + threadIdx.x;
    int k = (int)(i % Ktot);
    int n = (int)(i / Ktot);
    float v = (k < K0) ? Wx[(size_t)k * (4 * H_) + n]
                       : Wh[(size_t)(k - K0) * (4 * H_) + n];
    WT[i] = to_bf16(v);
}

__global__ __launch_bounds__(256) void convert_f32_bf16_kernel(const float* __restrict__ in,
                                                               __bf16* __restrict__ outp) {
    size_t i = (size_t)blockIdx.x * 256 + threadIdx.x;
    outp[i] = to_bf16(in[i]);
}

__global__ __launch_bounds__(256) void init_state_kernel(const float* __restrict__ h0,
                                                         const float* __restrict__ c0,
                                                         __bf16* __restrict__ h,
                                                         float* __restrict__ c) {
    size_t i = (size_t)blockIdx.x * 256 + threadIdx.x;
    h[i] = to_bf16(h0[i]);
    c[i] = c0[i];
}

// ---------------- fused LSTM step: gates GEMM (bf16 WMMA) + cell update ----------------
// gates(512 x 2048) = [A0 | A1](512 x (K0+K1)) @ WT^T + bias
// grid = (H/16 hidden tiles, B/32 batch tiles), block = 128 (4 waves, one gate each).
// Each wave owns TWO 16x16 accumulators (M and M+16) sharing one B fragment:
// 2 independent WMMAs per K-chunk fill each other's hazard slots.
template <int K0, int K1>
__global__ __launch_bounds__(128) void lstm_step_kernel(
    const __bf16* __restrict__ A0,              // input slab   (B x K0)
    const __bf16* __restrict__ A1,              // prev hidden  (B x K1)
    const __bf16* __restrict__ WT,              // (4H x (K0+K1)) bf16 row-major
    const float*  __restrict__ bias,            // (4H) f32
    float*        __restrict__ c,               // (B x H) f32, updated in place
    __bf16*       __restrict__ h_out)           // (B x H) bf16
{
    __shared__ float g_lds[4][32][17];

    const int lane = threadIdx.x & 31;
    const int gate = threadIdx.x >> 5;          // 0=i 1=f 2=g 3=o
    const int half = lane >> 4;
    const int l16  = lane & 15;
    const int mT = blockIdx.y, nT = blockIdx.x;
    const int brow  = mT * 32 + l16;            // first of two A rows this lane serves
    const int nfull = gate * H_ + nT * 16 + l16;
    constexpr int Ktot = K0 + K1;

    // Fixed per-lane base pointers; all K-loop loads become immediate offsets.
    const __bf16* a0lo = A0 + (size_t)brow * K0;
    const __bf16* a0hi = a0lo + (size_t)16 * K0;
    const __bf16* a1lo = A1 + (size_t)brow * K1;
    const __bf16* a1hi = a1lo + (size_t)16 * K1;
    const __bf16* wrow = WT + (size_t)nfull * Ktot;

    v8f acc0, acc1;
    {
        float bv = bias[nfull];                 // bias depends only on N column
        #pragma unroll
        for (int r = 0; r < 8; ++r) { acc0[r] = bv; acc1[r] = bv; }
    }

    #pragma unroll
    for (int kc = 0; kc < K0; kc += 32) {
        v16bf b  = load_b_frag(wrow + kc, half);
        v16bf aA = load_a_frag(a0lo + kc, half);
        v16bf aB = load_a_frag(a0hi + kc, half);
        acc0 = wmma_bf16(aA, b, acc0);
        acc1 = wmma_bf16(aB, b, acc1);
    }
    #pragma unroll
    for (int kc = 0; kc < K1; kc += 32) {
        v16bf b  = load_b_frag(wrow + K0 + kc, half);
        v16bf aA = load_a_frag(a1lo + kc, half);
        v16bf aB = load_a_frag(a1hi + kc, half);
        acc0 = wmma_bf16(aA, b, acc0);
        acc1 = wmma_bf16(aB, b, acc1);
    }

    // exchange the four gate tiles through LDS
    #pragma unroll
    for (int r = 0; r < 8; ++r) {
        g_lds[gate][r + 8 * half][l16]      = acc0[r];
        g_lds[gate][16 + r + 8 * half][l16] = acc1[r];
    }
    __syncthreads();

    // fused cell update: 512 elements, 128 threads
    #pragma unroll
    for (int e4 = 0; e4 < 4; ++e4) {
        int e = threadIdx.x + 128 * e4;
        int row = e >> 4, col = e & 15;
        float iv = fast_sigmoid(g_lds[0][row][col]);
        float fv = fast_sigmoid(g_lds[1][row][col]);
        float gv = fast_tanh   (g_lds[2][row][col]);
        float ov = fast_sigmoid(g_lds[3][row][col]);
        size_t idx = (size_t)(mT * 32 + row) * H_ + (size_t)(nT * 16 + col);
        float cn = fv * c[idx] + iv * gv;
        c[idx]     = cn;
        h_out[idx] = to_bf16(ov * fast_tanh(cn));
    }
}

// ---------------- fc1: (T*B, H) @ fc1_w^T (H2,H) + b -> out (B, T*H2) ----------------
// Same 2-accumulator scheme: each wave does rows [rowT*32, rowT*32+32) x one j-tile.
__global__ __launch_bounds__(128) void fc1_kernel(
    const __bf16* __restrict__ A,      // (T*B, H) bf16, row = t*B + b
    const __bf16* __restrict__ WTb,    // (H2, H) bf16
    const float*  __restrict__ bias,   // (H2)
    float*        __restrict__ out)    // (B, T*H2)
{
    const int lane = threadIdx.x & 31;
    const int wave = threadIdx.x >> 5;
    const int half = lane >> 4;
    const int l16  = lane & 15;
    const int rowT = blockIdx.x;
    const int jT   = blockIdx.y * 4 + wave;
    const int arow = rowT * 32 + l16;
    const int jcol = jT * 16 + l16;

    const __bf16* aplo = A + (size_t)arow * H_;
    const __bf16* aphi = aplo + (size_t)16 * H_;
    const __bf16* wp   = WTb + (size_t)jcol * H_;

    v8f acc0, acc1;
    {
        float bv = bias[jcol];
        #pragma unroll
        for (int r = 0; r < 8; ++r) { acc0[r] = bv; acc1[r] = bv; }
    }

    #pragma unroll
    for (int kc = 0; kc < H_; kc += 32) {
        v16bf b  = load_b_frag(wp + kc, half);
        v16bf aA = load_a_frag(aplo + kc, half);
        v16bf aB = load_a_frag(aphi + kc, half);
        acc0 = wmma_bf16(aA, b, acc0);
        acc1 = wmma_bf16(aB, b, acc1);
    }

    #pragma unroll
    for (int r = 0; r < 8; ++r) {
        int row0 = rowT * 32 + r + 8 * half;        // = t*B + b
        int row1 = row0 + 16;
        int col  = jT * 16 + l16;
        out[(size_t)(row0 & (B_ - 1)) * (NSEQ_ * H2_) + (size_t)(row0 >> 9) * H2_ + col] = acc0[r];
        out[(size_t)(row1 & (B_ - 1)) * (NSEQ_ * H2_) + (size_t)(row1 >> 9) * H2_ + col] = acc1[r];
    }
}

// ---------------- fc2: per-batch-row reduction (B, 32768) -> (B, 2) ----------------
__global__ __launch_bounds__(256) void fc2_kernel(const float* __restrict__ X,
                                                  const float* __restrict__ W,
                                                  const float* __restrict__ bias,
                                                  float* __restrict__ out) {
    __shared__ float red0[256], red1[256];
    const int b = blockIdx.x, tid = threadIdx.x;
    const int M = NSEQ_ * H2_;
    const float* x = X + (size_t)b * M;
    float s0 = 0.f, s1 = 0.f;
    for (int m = tid; m < M; m += 256) {
        float xv = x[m];
        s0 += xv * W[m];
        s1 += xv * W[M + m];
    }
    red0[tid] = s0; red1[tid] = s1;
    __syncthreads();
    for (int s = 128; s > 0; s >>= 1) {
        if (tid < s) { red0[tid] += red0[tid + s]; red1[tid] += red1[tid + s]; }
        __syncthreads();
    }
    if (tid == 0) {
        out[(size_t)b * 2 + 0] = red0[0] + bias[0];
        out[(size_t)b * 2 + 1] = red1[0] + bias[1];
    }
}

// ---------------- launch ----------------

extern "C" void kernel_launch(void* const* d_in, const int* in_sizes, int n_in,
                              void* d_out, int out_size, void* d_ws, size_t ws_size,
                              hipStream_t stream) {
    const float* x    = (const float*)d_in[0];
    const float* h0   = (const float*)d_in[1];
    const float* c0   = (const float*)d_in[2];
    const float* Wx1  = (const float*)d_in[3];
    const float* Wh1  = (const float*)d_in[4];
    const float* b1   = (const float*)d_in[5];
    const float* Wx2  = (const float*)d_in[6];
    const float* Wh2  = (const float*)d_in[7];
    const float* b2   = (const float*)d_in[8];
    const float* fc1w = (const float*)d_in[9];
    const float* fc1b = (const float*)d_in[10];
    const float* fc2w = (const float*)d_in[11];
    const float* fc2b = (const float*)d_in[12];
    float* out = (float*)d_out;
    (void)in_sizes; (void)n_in; (void)out_size; (void)ws_size;

    char* p = (char*)d_ws;
    auto carve = [&](size_t bytes) { char* r = p; p += (bytes + 255) & ~(size_t)255; return r; };

    __bf16* xT    = (__bf16*)carve((size_t)NSEQ_ * B_ * SEQ_ * 2);        // 32 MB
    __bf16* W1T   = (__bf16*)carve((size_t)4 * H_ * (SEQ_ + H_) * 2);     // 3 MB
    __bf16* W2T   = (__bf16*)carve((size_t)4 * H_ * (2 * H_) * 2);        // 4 MB
    __bf16* fc1wb = (__bf16*)carve((size_t)H2_ * H_ * 2);
    __bf16* h1buf = (__bf16*)carve((size_t)(NSEQ_ + 1) * B_ * H_ * 2);    // 64.5 MB
    __bf16* h2buf = (__bf16*)carve((size_t)(NSEQ_ + 1) * B_ * H_ * 2);    // 64.5 MB
    float*  cbuf  = (float*) carve((size_t)B_ * H_ * 4);
    float*  f1out = (float*) carve((size_t)B_ * NSEQ_ * H2_ * 4);         // 64 MB

    const size_t BH = (size_t)B_ * H_;

    convert_x_kernel<<<(NSEQ_ * B_ * SEQ_) / 256, 256, 0, stream>>>(x, xT);
    convert_wpair_kernel<<<(4 * H_ * (SEQ_ + H_)) / 256, 256, 0, stream>>>(
        Wx1, Wh1, W1T, SEQ_, SEQ_ + H_);
    convert_wpair_kernel<<<(4 * H_ * (2 * H_)) / 256, 256, 0, stream>>>(
        Wx2, Wh2, W2T, H_, 2 * H_);
    convert_f32_bf16_kernel<<<(H2_ * H_) / 256, 256, 0, stream>>>(fc1w, fc1wb);
    init_state_kernel<<<BH / 256, 256, 0, stream>>>(h0, c0, h1buf, cbuf);

    dim3 sgrid(H_ / 16, B_ / 32);   // 32 x 16 blocks, 4 waves each, M=32 per block

    // Layer 1: slot t+1 of h1buf gets h1[t]; cbuf updated in place.
    for (int t = 0; t < NSEQ_; ++t) {
        lstm_step_kernel<SEQ_, H_><<<sgrid, 128, 0, stream>>>(
            xT + (size_t)t * B_ * SEQ_,
            h1buf + (size_t)t * BH,
            W1T, b1, cbuf, h1buf + (size_t)(t + 1) * BH);
    }
    // Layer 2: initial h = h1 final (h1buf slot T), initial c = c1 final (cbuf as-is).
    for (int t = 0; t < NSEQ_; ++t) {
        const __bf16* prev_h = (t == 0) ? (h1buf + (size_t)NSEQ_ * BH)
                                        : (h2buf + (size_t)t * BH);
        lstm_step_kernel<H_, H_><<<sgrid, 128, 0, stream>>>(
            h1buf + (size_t)(t + 1) * BH,
            prev_h,
            W2T, b2, cbuf, h2buf + (size_t)(t + 1) * BH);
    }

    fc1_kernel<<<dim3((NSEQ_ * B_) / 32, (H2_ / 16) / 4), 128, 0, stream>>>(
        h2buf + BH, fc1wb, fc1b, f1out);
    fc2_kernel<<<B_, 256, 0, stream>>>(f1out, fc2w, fc2b, out);
}
